// ASCRM_7619271983683
// MI455X (gfx1250) — matrix operand order repaired and checked by reading, not compiled, not run on hardware
//
#include <hip/hip_runtime.h>
#include <hip/hip_fp16.h>
#include <hip/hip_bf16.h>

typedef __attribute__((ext_vector_type(16))) _Float16 v16h;
typedef __attribute__((ext_vector_type(8)))  _Float16 v8h;
typedef __attribute__((ext_vector_type(8)))  float    v8f;

#define WMMA_F16(a,b,c) __builtin_amdgcn_wmma_f32_16x16x32_f16(false,(a),false,(b),(short)0,(c),false,false)

// ---------------- WMMA fragment helpers (wave32, gfx1250; A/B use this code's contiguous k order, not the ISA table order) ----------------
// A 16x32 fragment: lane = half*16+m ; kh = half*16 ; a[j] = A[m][kh+j]
// B 32x16 fragment: lane = half*16+n ; kb = half*16 ; b[j] = B[kb+j][n]
// C 16x16 f32 frag: lane = half*16+n ; elem r = C[half*8+r][n]

// Contiguous (pre-swizzled) fragment load: 32 aligned bytes per lane.
__device__ inline v16h frag_ld16(const _Float16* p) { return *(const v16h*)p; }

// A fragment from row-major (MxK) storage; per-lane one aligned 32B load.
__device__ inline v16h frag_a(const _Float16* base, int ld) {
  int l = threadIdx.x & 31;
  int m = l & 15, kh = (l >> 4) << 4;
  const _Float16* r = base + m * ld;
  return *(const v16h*)(r + kh);
}
// B fragment where memory is (N x K) row-major (B transposed): contiguous per lane.
__device__ inline v16h frag_b_t(const _Float16* base, int ld) {
  int l = threadIdx.x & 31;
  int n = l & 15, kb = (l >> 4) << 4;
  return *(const v16h*)(base + n * ld + kb);
}
__device__ inline void store_c_f32(v8f c, float* base, int ld) {
  int l = threadIdx.x & 31;
  int n = l & 15, mb = (l >> 4) << 3;
#pragma unroll
  for (int r = 0; r < 8; ++r) base[(mb + r) * ld + n] = c[r];
}
// Store C fragment TRANSPOSED as f16 (dst is (N x M) row-major): 8 contiguous halves/lane.
__device__ inline void store_c_f16_t(v8f c, _Float16* base, int ld) {
  int l = threadIdx.x & 31;
  int n = l & 15, mb = (l >> 4) << 3;
  v8h pk;
#pragma unroll
  for (int r = 0; r < 8; ++r) pk[r] = (_Float16)c[r];
  *(v8h*)(base + n * ld + mb) = pk;
}

// ---------------- Kernel 1: per-patch attention block ----------------
// One workgroup (256 thr = 8 waves) per contributing "b" (ph2,pw2 in [0,18]).
#define NPH 19

__global__ __launch_bounds__(256) void attn_kernel(
    const float* __restrict__ exemplar, const float* __restrict__ query,
    const float* __restrict__ w_conv_e,
    _Float16* __restrict__ att_e, _Float16* __restrict__ att_q)
{
  extern __shared__ char smem_raw[];
  _Float16* sWs    = (_Float16*)smem_raw;      // 128x128 W_e, A-fragment swizzled
  _Float16* sEx    = sWs  + 128 * 128;         // 128x64 row-major (A use, GEMM3)
  _Float16* sQ     = sEx  + 128 * 64;          // 128x64 row-major (A use, GEMM4)
  _Float16* sExB   = sQ   + 128 * 64;          // 128x64 B-fragment swizzled (GEMM1)
  _Float16* sQB    = sExB + 128 * 64;          // 128x64 B-fragment swizzled (GEMM2)
  _Float16* sCorrT = sQB  + 128 * 64;          // 64x128 Corr^T (d-major) f16
  _Float16* sA1B   = sCorrT + 64 * 128;        // 64x64 A1, B-fragment swizzled (GEMM3)
  _Float16* sB2    = sA1B + 64 * 64;           // 64x64 B2 row-major (B^T use, GEMM4)
  float*    sA     = (float*)(sB2 + 64 * 64);  // 64x64 logits

  const int tid = threadIdx.x;
  const int wid = tid >> 5;
  const int l   = tid & 31;

  int bidx = blockIdx.x;
  int n2  = bidx / (NPH * NPH);
  int pr  = bidx % (NPH * NPH);
  int ph2 = pr / NPH, pw2 = pr % NPH;
  int b   = n2 * 961 + ph2 * 31 + pw2;          // flat "b" of reference reshape

  // ---- stage W_e straight into A-fragment layout: sWs[((wid*4+ks)*32+lane)*16+j]
  for (int i = tid; i < 128 * 128; i += 256) {
    int j = i & 15, lane = (i >> 4) & 31, ks = (i >> 9) & 3, wt = i >> 11;
    int o  = wt * 16 + (lane & 15);
    int kk = ((lane >> 4) << 4) + j;
    sWs[i] = (_Float16)w_conv_e[o * 128 + ks * 32 + kk];
  }

  // ---- stage Ex/Q: one thread per (tensor, row c2); decode mapping once.
  {
    int tensor = tid >> 7;            // 0 = exemplar, 1 = query
    int c2 = tid & 127;
    int g   = b * 128 + c2;
    int n   = g / 123008;             // 123008 = 128*961
    int rem = g % 123008;
    int c   = rem / 961;
    int L   = rem % 961;
    int ph  = L / 31, pw = L % 31;
    const float* src = tensor ? query : exemplar;
    _Float16* rowA = (tensor ? sQ : sEx) + c2 * 64;
    _Float16* dstB = tensor ? sQB : sExB;
    int base = ((n * 128 + c) << 14) + ((ph * 4) << 7) + (pw * 4);
    int ksr = c2 >> 5, half = (c2 >> 4) & 1, jr = c2 & 15;  // B-frag coords for k=c2
    for (int d = 0; d < 64; ++d) {
      _Float16 v = (_Float16)0.f;
      if (d < 49) {
        int kh = d / 7, kw = d % 7;
        v = (_Float16)src[base + (kh << 7) + kw];
      }
      rowA[d] = v;
      int dt = d >> 4, nn = d & 15;
      dstB[(((ksr * 4 + dt) * 32) + half * 16 + nn) * 16 + jr] = v;
    }
  }
  __syncthreads();

  // ---- GEMM1: Corr(128x64) = W(128x128) * Ex(128x64); store Corr^T (64x128)
  {
    v8f acc[4] = {v8f{}, v8f{}, v8f{}, v8f{}};
#pragma unroll
    for (int ks = 0; ks < 4; ++ks) {
      v16h a = frag_ld16(sWs + ((wid * 4 + ks) * 32 + l) * 16);
#pragma unroll
      for (int dt = 0; dt < 4; ++dt) {
        v16h bb = frag_ld16(sExB + ((ks * 4 + dt) * 32 + l) * 16);
        acc[dt] = WMMA_F16(a, bb, acc[dt]);
      }
    }
#pragma unroll
    for (int dt = 0; dt < 4; ++dt)
      store_c_f16_t(acc[dt], sCorrT + (dt * 16) * 128 + wid * 16, 128);
  }
  __syncthreads();

  // ---- GEMM2: A(64x64) = Corr^T(64x128) * Q(128x64); 16 tiles over 8 waves
  {
    int dt = wid >> 1;
    int e0 = (wid & 1) * 2;
    v8f acc[2] = {v8f{}, v8f{}};
#pragma unroll
    for (int ks = 0; ks < 4; ++ks) {
      v16h a = frag_a(sCorrT + dt * 16 * 128 + ks * 32, 128);
#pragma unroll
      for (int i = 0; i < 2; ++i) {
        v16h bb = frag_ld16(sQB + ((ks * 4 + e0 + i) * 32 + l) * 16);
        acc[i] = WMMA_F16(a, bb, acc[i]);
      }
    }
#pragma unroll
    for (int i = 0; i < 2; ++i) store_c_f32(acc[i], sA + dt * 16 * 64 + (e0 + i) * 16, 64);
  }
  __syncthreads();

  // ---- Softmaxes. A1 (over d) written in B-fragment layout; B2 (over e) row-major.
  if (tid < 64) {
    int e = tid, et = e >> 4, n = e & 15;
    float m = -1e30f;
    for (int d = 0; d < 49; ++d) m = fmaxf(m, sA[d * 64 + e]);
    float s = 0.f;
    for (int d = 0; d < 49; ++d) s += __expf(sA[d * 64 + e] - m);
    float inv = 1.f / s;
    for (int d = 0; d < 64; ++d) {
      float v = (d < 49) ? __expf(sA[d * 64 + e] - m) * inv : 0.f;
      int ks = d >> 5, half = (d >> 4) & 1, j = d & 15;
      sA1B[((ks * 4 + et) * 32 + half * 16 + n) * 16 + j] = (_Float16)v;
    }
  } else if (tid < 128) {
    int d = tid - 64;
    if (d < 49) {
      float m = -1e30f;
      for (int e = 0; e < 49; ++e) m = fmaxf(m, sA[d * 64 + e]);
      float s = 0.f;
      for (int e = 0; e < 49; ++e) s += __expf(sA[d * 64 + e] - m);
      float inv = 1.f / s;
      for (int e = 0; e < 49; ++e) sB2[d * 64 + e] = (_Float16)(__expf(sA[d * 64 + e] - m) * inv);
      for (int e = 49; e < 64; ++e) sB2[d * 64 + e] = (_Float16)0.f;
    } else {
      for (int e = 0; e < 64; ++e) sB2[d * 64 + e] = (_Float16)0.f;
    }
  }
  __syncthreads();

  // ---- GEMM3: QAtt = Ex * A1 ; GEMM4: EAtt = Q * B2^T ; recon-scatter to global.
  {
    v8f acc3[4] = {v8f{}, v8f{}, v8f{}, v8f{}};
    v8f acc4[4] = {v8f{}, v8f{}, v8f{}, v8f{}};
#pragma unroll
    for (int ks = 0; ks < 2; ++ks) {
      v16h a3 = frag_a(sEx + wid * 16 * 64 + ks * 32, 64);
      v16h a4 = frag_a(sQ  + wid * 16 * 64 + ks * 32, 64);
#pragma unroll
      for (int t = 0; t < 4; ++t) {
        v16h b3 = frag_ld16(sA1B + ((ks * 4 + t) * 32 + l) * 16);
        acc3[t] = WMMA_F16(a3, b3, acc3[t]);
        v16h b4 = frag_b_t(sB2 + t * 16 * 64 + ks * 32, 64);
        acc4[t] = WMMA_F16(a4, b4, acc4[t]);
      }
    }
    int nn = l & 15, mb = (l >> 4) << 3;
#pragma unroll
    for (int t = 0; t < 4; ++t) {
      int col = t * 16 + nn;                    // e (QAtt) or d (EAtt): same recon map
      if (col < 49) {
        int kh = col / 7, kw = col % 7;
        int h = ph2 * 7 + kh, w = pw2 * 7 + kw;
        if (h < 128 && w < 128) {
#pragma unroll
          for (int r = 0; r < 8; ++r) {
            int cc = wid * 16 + mb + r;
            int o = ((n2 * 128 + cc) << 14) + (h << 7) + w;
            att_q[o] = (_Float16)acc3[t][r];
            att_e[o] = (_Float16)acc4[t][r];
          }
        }
      }
    }
  }
}

// ---------------- Kernel 2: sigmoid gate * att + residual ----------------
__global__ __launch_bounds__(256) void gate_residual_kernel(
    const _Float16* __restrict__ att, const float* __restrict__ orig,
    const float* __restrict__ wg, _Float16* __restrict__ z)
{
  int pi = blockIdx.x * 256 + threadIdx.x;   // < 8*16384 pixels
  int n = pi >> 14, p = pi & 16383;
  int base = (n * 128) << 14;
  float s = 0.f;
  for (int c = 0; c < 128; ++c) s += wg[c] * (float)att[base + (c << 14) + p];
  float mask = 1.f / (1.f + __expf(-s));
  for (int c = 0; c < 128; ++c) {
    int idx = base + (c << 14) + p;
    z[idx] = (_Float16)((float)att[idx] * mask + orig[idx]);
  }
}

// ---------------- Kernel 3: depthwise 3x3 + BN + ReLU ----------------
__global__ __launch_bounds__(256) void dw_bn_relu_kernel(
    const _Float16* __restrict__ in, const float* __restrict__ wdw,
    const float* __restrict__ g, const float* __restrict__ bb,
    _Float16* __restrict__ out, int CH)
{
  int gid = blockIdx.x * 256 + threadIdx.x;
  int w = gid & 127, h = (gid >> 7) & 127;
  int nc = gid >> 14;
  int c = nc % CH;
  const _Float16* pin = in + (nc << 14);
  const float* wk = wdw + c * 9;
  float acc = 0.f;
#pragma unroll
  for (int dh = -1; dh <= 1; ++dh) {
    int hh = h + dh;
    if (hh < 0 || hh > 127) continue;
#pragma unroll
    for (int dx = -1; dx <= 1; ++dx) {
      int ww = w + dx;
      if (ww < 0 || ww > 127) continue;
      acc += wk[(dh + 1) * 3 + (dx + 1)] * (float)pin[(hh << 7) + ww];
    }
  }
  float scale = g[c] * rsqrtf(1.0f + 1e-5f);
  out[gid] = (_Float16)fmaxf(0.f, scale * acc + bb[c]);
}

// ---------------- Kernel 4: pointwise conv as WMMA GEMM + BN + ReLU ----------------
// Per block: 128 pixels of one image; wave wid owns 16 pixels, all 8 o-tiles.
template <typename OutT, int CIN>
__global__ __launch_bounds__(256) void pw_bn_relu_kernel(
    const _Float16* __restrict__ in, const float* __restrict__ wpw,
    const float* __restrict__ g, const float* __restrict__ bb,
    OutT* __restrict__ out, int OCH_TOT, int o_off)
{
  constexpr int NK = CIN / 32;
  extern __shared__ char smem_raw[];
  _Float16* sWp = (_Float16*)smem_raw;          // 128 x CIN, A-fragment swizzled
  const int tid = threadIdx.x, wid = tid >> 5, l = tid & 31;

  // stage weights straight into A-fragment layout: sWp[((ot*NK+ks)*32+lane)*16+j]
  for (int i = tid; i < 128 * CIN; i += 256) {
    int j = i & 15, lane = (i >> 4) & 31;
    int rest = i >> 9;
    int ks = rest % NK, ot = rest / NK;
    int o  = ot * 16 + (lane & 15);
    int kk = ((lane >> 4) << 4) + j;
    sWp[i] = (_Float16)wpw[o * CIN + ks * 32 + kk];
  }
  __syncthreads();

  int bid = blockIdx.x;
  int n = bid >> 7;
  int pbase = (bid & 127) * 128 + wid * 16;
  const _Float16* pin = in + ((n * CIN) << 14) + pbase;

  v8f acc[8] = {v8f{}, v8f{}, v8f{}, v8f{}, v8f{}, v8f{}, v8f{}, v8f{}};
  int kb = (l >> 4) << 4, npx = l & 15;
#pragma unroll
  for (int ks = 0; ks < NK; ++ks) {
    if (ks + 1 < NK)
      __builtin_prefetch(pin + (((ks + 1) * 32 + kb) << 14) + npx, 0, 1);
    v16h bfrag;
#pragma unroll
    for (int j = 0; j < 16; ++j)
      bfrag[j] = pin[((ks * 32 + kb + j) << 14) + npx];
#pragma unroll
    for (int ot = 0; ot < 8; ++ot) {
      v16h a = frag_ld16(sWp + ((ot * NK + ks) * 32 + l) * 16);
      acc[ot] = WMMA_F16(a, bfrag, acc[ot]);
    }
  }
  float sc = rsqrtf(1.0f + 1e-5f);
  int mb = (l >> 4) << 3;
#pragma unroll
  for (int ot = 0; ot < 8; ++ot) {
#pragma unroll
    for (int r = 0; r < 8; ++r) {
      int o = ot * 16 + mb + r;
      float v = fmaxf(0.f, g[o] * sc * acc[ot][r] + bb[o]);
      out[((n * OCH_TOT + o_off + o) << 14) + pbase + npx] = (OutT)v;
    }
  }
}

// ---------------- Host launcher ----------------
extern "C" void kernel_launch(void* const* d_in, const int* in_sizes, int n_in,
                              void* d_out, int out_size, void* d_ws, size_t ws_size,
                              hipStream_t stream)
{
  (void)in_sizes; (void)n_in; (void)out_size; (void)ws_size;
  const float* exemplar = (const float*)d_in[0];
  const float* query    = (const float*)d_in[1];
  const float* w_conv_e = (const float*)d_in[2];
  const float* w_gate1  = (const float*)d_in[3];
  const float* w_gate2  = (const float*)d_in[4];
  const float* dw1_w    = (const float*)d_in[5];
  const float* bn1a_g   = (const float*)d_in[6];
  const float* bn1a_b   = (const float*)d_in[7];
  const float* pw1_w    = (const float*)d_in[8];
  const float* bn1b_g   = (const float*)d_in[9];
  const float* bn1b_b   = (const float*)d_in[10];
  const float* dwf_w    = (const float*)d_in[11];
  const float* bnfa_g   = (const float*)d_in[12];
  const float* bnfa_b   = (const float*)d_in[13];
  const float* pwf_w    = (const float*)d_in[14];
  const float* bnfb_g   = (const float*)d_in[15];
  const float* bnfb_b   = (const float*)d_in[16];
  float* out = (float*)d_out;

  const size_t E = (size_t)8 * 128 * 128 * 128;   // 16,777,216 elements
  _Float16* ws = (_Float16*)d_ws;                 // total use: 4*E halves = 128 MiB
  _Float16* att_e = ws;
  _Float16* att_q = ws + E;
  _Float16* z_e   = ws + 2 * E;
  _Float16* z_q   = ws + 3 * E;
  _Float16* t_e   = att_e;          // att dead after gate
  _Float16* t_q   = att_q;
  _Float16* cat   = z_e;            // (8,256,16384); z dead after dw1
  _Float16* tf    = att_e;          // (8,256,16384); t dead after pw1

  // sWs + sEx + sQ + sExB + sQB (5*8192 + 8192 extra for sWs=16384) + sCorrT + sA1B + sB2, + sA f32
  size_t attn_lds = (size_t)(128*128 + 4*128*64 + 64*128 + 2*64*64) * sizeof(_Float16)
                  + (size_t)(64 * 64) * sizeof(float);   // 147,456 B

  attn_kernel<<<dim3(8 * NPH * NPH), dim3(256), attn_lds, stream>>>(
      exemplar, query, w_conv_e, att_e, att_q);

  gate_residual_kernel<<<512, 256, 0, stream>>>(att_e, exemplar, w_gate1, z_e);
  gate_residual_kernel<<<512, 256, 0, stream>>>(att_q, query,    w_gate2, z_q);

  dw_bn_relu_kernel<<<65536, 256, 0, stream>>>(z_e, dw1_w, bn1a_g, bn1a_b, t_e, 128);
  dw_bn_relu_kernel<<<65536, 256, 0, stream>>>(z_q, dw1_w, bn1a_g, bn1a_b, t_q, 128);

  pw_bn_relu_kernel<_Float16, 128><<<1024, 256, 128 * 128 * sizeof(_Float16), stream>>>(
      t_e, pw1_w, bn1b_g, bn1b_b, cat, 256, 0);
  pw_bn_relu_kernel<_Float16, 128><<<1024, 256, 128 * 128 * sizeof(_Float16), stream>>>(
      t_q, pw1_w, bn1b_g, bn1b_b, cat, 256, 128);

  dw_bn_relu_kernel<<<131072, 256, 0, stream>>>(cat, dwf_w, bnfa_g, bnfa_b, tf, 256);

  pw_bn_relu_kernel<float, 256><<<1024, 256, 128 * 256 * sizeof(_Float16), stream>>>(
      tf, pwf_w, bnfb_g, bnfb_b, out, 128, 0);
}
